// LinearAttentionLayerOptimized_63153199121102
// MI455X (gfx1250) — matrix operand order repaired and checked by reading, not compiled, not run on hardware
//
#include <hip/hip_runtime.h>
#include <math.h>

// ---------------- types ----------------
typedef __bf16 bf16_t;
typedef __attribute__((ext_vector_type(16))) __bf16 v16bf;
typedef __attribute__((ext_vector_type(8)))  __bf16 v8bf;
typedef __attribute__((ext_vector_type(8)))  float  v8f;

#define BM 128
#define BN 128
#define BK 32
#define LDS_STRIDE 40   // bf16 elems; 80B row stride keeps 16B alignment, skews banks

enum { EPI_GELU_BF16 = 0, EPI_BIAS_F32 = 1, EPI_ID_BF16 = 2, EPI_SIG_F32 = 3, EPI_DECAY_F32 = 4 };

__device__ __forceinline__ float gelu_exact(float x) {
    return 0.5f * x * (1.0f + erff(x * 0.7071067811865476f));
}
__device__ __forceinline__ float sigmoidf_(float x) {
    return 1.0f / (1.0f + expf(-x));
}

// ---------------- small prep kernels ----------------
__global__ void cast_f32_to_bf16_kernel(const float* __restrict__ src, bf16_t* __restrict__ dst, int n) {
    int i = blockIdx.x * blockDim.x + threadIdx.x;
    if (i < n) dst[i] = (bf16_t)src[i];
}

// dst[R2][C2] bf16 = zero-padded src[R1][C1] fp32
__global__ void pad_convert_kernel(const float* __restrict__ src, bf16_t* __restrict__ dst,
                                   int R1, int C1, int R2, int C2) {
    long long i = (long long)blockIdx.x * blockDim.x + threadIdx.x;
    long long tot = (long long)R2 * C2;
    if (i >= tot) return;
    int r = (int)(i / C2), c = (int)(i % C2);
    float v = (r < R1 && c < C1) ? src[(long long)r * C1 + c] : 0.0f;
    dst[i] = (bf16_t)v;
}

// out[e] = (e < E) ? b[e] + sum_c mu[c]*W[c*E+e]  : 0      (mu/W/b each optional)
__global__ void bias_fold_kernel(const float* __restrict__ mu, const float* __restrict__ W,
                                 const float* __restrict__ b, float* __restrict__ out,
                                 int C, int E, int Epad) {
    int e = blockIdx.x * blockDim.x + threadIdx.x;
    if (e >= Epad) return;
    float s = 0.0f;
    if (e < E) {
        if (b) s = b[e];
        if (mu && W) {
            for (int c = 0; c < C; ++c) s += mu[c] * W[(long long)c * E + e];
        }
    }
    out[e] = s;
}

__global__ void fill_zero_kernel(float* __restrict__ p, int n) {
    int i = blockIdx.x * blockDim.x + threadIdx.x;
    if (i < n) p[i] = 0.0f;
}

// ---------------- WMMA GEMM:  C = epi(A[M,K] @ B[K,N] + bias[N]) ----------------
// A, B bf16 row-major; fp32 accum via v_wmma_f32_16x16x32_bf16.
// Block tile 128x128, 256 threads = 8 waves (4 M-groups x 2 N-groups),
// each wave owns a 2x4 grid of 16x16 tiles -> 8 WMMAs per K-step off 6 fragments.
// Double-buffered LDS: global loads for K-step k+1 overlap the WMMAs of step k.
// Requires M%128==0, N%128==0, K%32==0. bias must be non-null (length N).
template <int EPI>
__global__ __launch_bounds__(256)
void gemm_bf16_kernel(const bf16_t* __restrict__ A, const bf16_t* __restrict__ Bm,
                      const float* __restrict__ bias, void* __restrict__ Cout,
                      int M, int N, int K) {
    __shared__ __align__(16) bf16_t lA[2][BM * LDS_STRIDE];   // A tile  [m][k]
    __shared__ __align__(16) bf16_t lB[2][BN * LDS_STRIDE];   // B tile transposed [n][k]

    const int tid  = threadIdx.x;
    const int lane = tid & 31;
    const int wave = tid >> 5;            // 0..7
    const int wr   = wave & 3;            // M tile pair {2*wr, 2*wr+1}
    const int wc   = wave >> 2;           // N tile quad {4*wc .. 4*wc+3}
    const int m0 = blockIdx.y * BM;
    const int n0 = blockIdx.x * BN;

    // staging maps: A 128x32 -> 16 bf16/thread ; B 32x128 -> 2 k-rows x 8 n /thread
    const int ar = tid >> 1;              // 0..127
    const int ac = (tid & 1) * 16;        // 0 or 16
    const int bk = (tid >> 4) * 2;        // 0,2,...,30
    const int bn = (tid & 15) * 8;        // 0,8,...,120

    const bf16_t* gA = A + (long long)(m0 + ar) * K + ac;
    const bf16_t* gB = Bm + (long long)bk * N + n0 + bn;

    v8bf ra0, ra1, rb0, rb1;
    // prologue: load K-step 0
    ra0 = *(const v8bf*)(gA);
    ra1 = *(const v8bf*)(gA + 8);
    rb0 = *(const v8bf*)(gB);
    rb1 = *(const v8bf*)(gB + N);
    gA += BK; gB += (long long)BK * N;

    auto stage = [&](int buf) {
        *(v8bf*)&lA[buf][ar * LDS_STRIDE + ac]     = ra0;
        *(v8bf*)&lA[buf][ar * LDS_STRIDE + ac + 8] = ra1;
        const unsigned short* p0 = (const unsigned short*)&rb0;
        const unsigned short* p1 = (const unsigned short*)&rb1;
#pragma unroll
        for (int i = 0; i < 8; ++i) {
            unsigned u = (unsigned)p0[i] | ((unsigned)p1[i] << 16);   // {k, k+1} pair
            *(unsigned*)&lB[buf][(bn + i) * LDS_STRIDE + bk] = u;
        }
    };
    stage(0);

    v8f acc[2][4] = {};
    const int KT = K / BK;
    int cur = 0;
    for (int ks = 0; ks < KT; ++ks) {
        __syncthreads();                      // buf[cur] ready for all waves
        const bool more = (ks + 1 < KT);
        if (more) {                            // issue next tile's global loads (overlap)
            ra0 = *(const v8bf*)(gA);
            ra1 = *(const v8bf*)(gA + 8);
            rb0 = *(const v8bf*)(gB);
            rb1 = *(const v8bf*)(gB + N);
            gA += BK; gB += (long long)BK * N;
        }

        // ---- fragments from buf[cur]
        // A: lane<16 -> row l, K {0..7,16..23}; lane>=16 -> K {8..15,24..31}
        // B: lane<16 -> col l, K {0..7,16..23}; lane>=16 -> K {8..15,24..31}
        const int klo = (lane < 16) ? 0 : 8;
        const int kb  = (lane < 16) ? 0 : 8;
        v16bf af[2], bfr[4];
#pragma unroll
        for (int i = 0; i < 2; ++i) {
            int arow = 16 * (2 * wr + i) + (lane & 15);
            union { v16bf v; v8bf h[2]; } ua;
            ua.h[0] = *(const v8bf*)&lA[cur][arow * LDS_STRIDE + klo];
            ua.h[1] = *(const v8bf*)&lA[cur][arow * LDS_STRIDE + klo + 16];
            af[i] = ua.v;
        }
#pragma unroll
        for (int i = 0; i < 4; ++i) {
            int bcol = 16 * (4 * wc + i) + (lane & 15);
            union { v16bf v; v8bf h[2]; } ub;
            ub.h[0] = *(const v8bf*)&lB[cur][bcol * LDS_STRIDE + kb];
            ub.h[1] = *(const v8bf*)&lB[cur][bcol * LDS_STRIDE + kb + 16];
            bfr[i] = ub.v;
        }
#pragma unroll
        for (int mi = 0; mi < 2; ++mi)
#pragma unroll
            for (int ni = 0; ni < 4; ++ni)
                acc[mi][ni] = __builtin_amdgcn_wmma_f32_16x16x32_bf16(
                    false, af[mi], false, bfr[ni], (short)0, acc[mi][ni], false, false);

        if (more) stage(cur ^ 1);             // waits on loads after WMMAs issued
        cur ^= 1;
    }

    // ---- epilogue: lane l, reg r -> row = 16*tileM + r + 8*(l>=16), col = 16*tileN + (l&15)
    const int rbase = (lane >> 4) * 8;
#pragma unroll
    for (int mi = 0; mi < 2; ++mi) {
#pragma unroll
        for (int ni = 0; ni < 4; ++ni) {
            const v8f& acc2 = acc[mi][ni];
            int gn = n0 + 16 * (4 * wc + ni) + (lane & 15);
            float bv = bias[gn];
#pragma unroll
            for (int r = 0; r < 8; ++r) {
                int gm = m0 + 16 * (2 * wr + mi) + rbase + r;
                float x = acc2[r] + bv;
                long long idx = (long long)gm * N + gn;
                if constexpr (EPI == EPI_GELU_BF16) {
                    ((bf16_t*)Cout)[idx] = (bf16_t)gelu_exact(x);
                } else if constexpr (EPI == EPI_ID_BF16) {
                    ((bf16_t*)Cout)[idx] = (bf16_t)x;
                } else if constexpr (EPI == EPI_BIAS_F32) {
                    ((float*)Cout)[idx] = x;
                } else if constexpr (EPI == EPI_SIG_F32) {
                    ((float*)Cout)[idx] = sigmoidf_(x);
                } else { // EPI_DECAY_F32 : exp(-c * sigmoid(tanh(x)))
                    ((float*)Cout)[idx] = expf(-0.7408182206817179f * sigmoidf_(tanhf(x)));
                }
            }
        }
    }
}

// ---------------- sequential rank-1 state recurrence ----------------
// One 64-thread block per (b,h). Thread t owns row t of S and row t of S^T.
__global__ __launch_bounds__(64)
void recurrence_kernel(const float* __restrict__ R, const float* __restrict__ Kx,
                       const float* __restrict__ V, const float* __restrict__ Iclr,
                       const float* __restrict__ Decay,
                       const float* __restrict__ rem_mult, const float* __restrict__ iclr_mix,
                       const float* __restrict__ bonus_mult, float* __restrict__ Out) {
    const int bh = blockIdx.x;          // 0..63
    const int b = bh >> 4, h = bh & 15;
    const int t0 = threadIdx.x;         // 0..63
    const int cbase = h * 64;
    const float remf = rem_mult[cbase + t0];
    const float mixf = iclr_mix[cbase + t0];
    const float bmf  = bonus_mult[cbase + t0];

    __shared__ float sd[64], sc[64], skn[64], srk[64], sv[64], sr[64], stmp[64], red[128];

    float S[64], ST[64];
#pragma unroll
    for (int i = 0; i < 64; ++i) { S[i] = 0.0f; ST[i] = 0.0f; }

    const long long base0 = ((long long)b * 2048) * 1024 + cbase;
    for (int t = 0; t < 2048; ++t) {
        long long idx = base0 + (long long)t * 1024 + t0;
        float kk = Kx[idx], rr = R[idx], vv = V[idx];
        float aa = Iclr[idx], dd = Decay[idx];
        float rm = kk * remf;
        float rk = kk * (1.0f + (aa - 1.0f) * mixf);

        red[t0]      = rm * rm;   // for ||removal_k||
        red[64 + t0] = rr * rk;   // for bonus dot
        __syncthreads();
        float s1 = 0.0f, s2 = 0.0f;
#pragma unroll
        for (int i = 0; i < 64; ++i) { s1 += red[i]; s2 += red[64 + i]; }
        float kn = rm / fmaxf(sqrtf(s1), 1e-12f);
        float bonus = 0.2f * s2;
        __syncthreads();

        sd[t0] = dd; sc[t0] = aa * kn; skn[t0] = kn;
        srk[t0] = rk; sv[t0] = vv; sr[t0] = rr;
        __syncthreads();

        // tmp[t0] = sum_j (a*kn)[j] * S[j][t0]   (lane-local via S^T row)
        float tmp = 0.0f;
#pragma unroll
        for (int j = 0; j < 64; ++j) tmp += sc[j] * ST[j];
        stmp[t0] = tmp;
        __syncthreads();

        // S row update + output dot (lane-local)
        float out = 0.0f;
#pragma unroll
        for (int k2 = 0; k2 < 64; ++k2) {
            float s = dd * S[k2] - kn * stmp[k2] + vv * srk[k2];
            S[k2] = s;
            out += s * sr[k2];
        }
        // S^T row update: ST[j] = d[j]*ST[j] - kn[j]*tmp_{t0} + v[j]*rk_{t0}
#pragma unroll
        for (int j = 0; j < 64; ++j)
            ST[j] = sd[j] * ST[j] - skn[j] * tmp + sv[j] * rk;

        Out[idx] = out + bonus * vv * bmf;
        __syncthreads();
    }
}

// ---------------- LayerNorm * gate -> bf16 ----------------
__global__ __launch_bounds__(256)
void ln_gate_kernel(const float* __restrict__ X, const float* __restrict__ gate,
                    const float* __restrict__ g, const float* __restrict__ beta,
                    bf16_t* __restrict__ Y) {
    const int row = blockIdx.x;
    const float* x = X + (long long)row * 1024;
    __shared__ float red[256];
    float local[4];
    float s = 0.0f;
#pragma unroll
    for (int i = 0; i < 4; ++i) { local[i] = x[threadIdx.x + 256 * i]; s += local[i]; }
    red[threadIdx.x] = s;
    __syncthreads();
    for (int off = 128; off > 0; off >>= 1) {
        if (threadIdx.x < off) red[threadIdx.x] += red[threadIdx.x + off];
        __syncthreads();
    }
    float mean = red[0] * (1.0f / 1024.0f);
    __syncthreads();
    float v = 0.0f;
#pragma unroll
    for (int i = 0; i < 4; ++i) { float d = local[i] - mean; v += d * d; }
    red[threadIdx.x] = v;
    __syncthreads();
    for (int off = 128; off > 0; off >>= 1) {
        if (threadIdx.x < off) red[threadIdx.x] += red[threadIdx.x + off];
        __syncthreads();
    }
    float inv = rsqrtf(red[0] * (1.0f / 1024.0f) + 1e-6f);
#pragma unroll
    for (int i = 0; i < 4; ++i) {
        int c = threadIdx.x + 256 * i;
        float y = (local[i] - mean) * inv * g[c] + beta[c];
        y *= gate[(long long)row * 1024 + c];
        Y[(long long)row * 1024 + c] = (bf16_t)y;
    }
}

// ---------------- host-side orchestration ----------------
extern "C" void kernel_launch(void* const* d_in, const int* in_sizes, int n_in,
                              void* d_out, int out_size, void* d_ws, size_t ws_size,
                              hipStream_t stream) {
    (void)in_sizes; (void)n_in; (void)out_size; (void)ws_size;
    const int Mtok = 8192, C = 1024, E = 1228, Ep = 1280, L = 128;

    const float* q     = (const float*)d_in[0];
    const float* mu_r  = (const float*)d_in[4];
    const float* mu_k  = (const float*)d_in[5];
    const float* mu_v  = (const float*)d_in[6];
    const float* mu_g  = (const float*)d_in[7];
    const float* mu_a  = (const float*)d_in[8];
    const float* mu_d  = (const float*)d_in[9];
    const float* dA = (const float*)d_in[10]; const float* dB = (const float*)d_in[11];
    const float* db = (const float*)d_in[12];
    const float* aA = (const float*)d_in[13]; const float* aB = (const float*)d_in[14];
    const float* ab = (const float*)d_in[15];
    const float* gA = (const float*)d_in[16]; const float* gB = (const float*)d_in[17];
    const float* W1f[4] = { (const float*)d_in[18], (const float*)d_in[22],
                            (const float*)d_in[26], (const float*)d_in[30] };
    const float* b1f[4] = { (const float*)d_in[19], (const float*)d_in[23],
                            (const float*)d_in[27], (const float*)d_in[31] };
    const float* W2f[4] = { (const float*)d_in[20], (const float*)d_in[24],
                            (const float*)d_in[28], (const float*)d_in[32] };
    const float* b2f[4] = { (const float*)d_in[21], (const float*)d_in[25],
                            (const float*)d_in[29], (const float*)d_in[33] };
    const float* rem_mult   = (const float*)d_in[34];
    const float* iclr_mix   = (const float*)d_in[35];
    const float* bonus_mult = (const float*)d_in[36];
    const float* ln_g = (const float*)d_in[37];
    const float* ln_b = (const float*)d_in[38];
    const float* muMLP[4] = { mu_r, mu_k, mu_v, nullptr };

    // ---- workspace carve-out
    char* p = (char*)d_ws;
    auto alloc = [&](size_t bytes) -> void* {
        void* r = (void*)p; p += (bytes + 255) & ~(size_t)255; return r;
    };
    bf16_t* Xbf = (bf16_t*)alloc((size_t)Mtok * C * 2);
    bf16_t* H   = (bf16_t*)alloc((size_t)Mtok * Ep * 2);
    bf16_t* W1p[4]; for (int i = 0; i < 4; ++i) W1p[i] = (bf16_t*)alloc((size_t)C * Ep * 2);
    bf16_t* W2p[4]; for (int i = 0; i < 4; ++i) W2p[i] = (bf16_t*)alloc((size_t)Ep * C * 2);
    bf16_t* LAc[3]; for (int i = 0; i < 3; ++i) LAc[i] = (bf16_t*)alloc((size_t)C * L * 2);
    bf16_t* LBc[3]; for (int i = 0; i < 3; ++i) LBc[i] = (bf16_t*)alloc((size_t)L * C * 2);
    float* b1adj[4]; for (int i = 0; i < 4; ++i) b1adj[i] = (float*)alloc((size_t)Ep * 4);
    float* bL[3];    for (int i = 0; i < 3; ++i) bL[i]    = (float*)alloc((size_t)L * 4);
    float* zbias = (float*)alloc((size_t)Ep * 4);
    float* Rf    = (float*)alloc((size_t)Mtok * C * 4);
    float* Kf    = (float*)alloc((size_t)Mtok * C * 4);
    float* Vf    = (float*)alloc((size_t)Mtok * C * 4);
    float* Iclr  = (float*)alloc((size_t)Mtok * C * 4);
    float* Decay = (float*)alloc((size_t)Mtok * C * 4);
    float* Gate  = (float*)alloc((size_t)Mtok * C * 4);
    float* RecO  = (float*)alloc((size_t)Mtok * C * 4);

    // ---- prep
    {
        int n = Mtok * C;
        cast_f32_to_bf16_kernel<<<(n + 255) / 256, 256, 0, stream>>>(q, Xbf, n);
    }
    for (int i = 0; i < 4; ++i) {
        long long t1 = (long long)C * Ep;
        pad_convert_kernel<<<(unsigned)((t1 + 255) / 256), 256, 0, stream>>>(W1f[i], W1p[i], C, E, C, Ep);
        long long t2 = (long long)Ep * C;
        pad_convert_kernel<<<(unsigned)((t2 + 255) / 256), 256, 0, stream>>>(W2f[i], W2p[i], E, C, Ep, C);
    }
    const float* LAsrc[3] = { gA, aA, dA };
    const float* LBsrc[3] = { gB, aB, dB };
    const float* muL[3]   = { mu_g, mu_a, mu_d };
    for (int i = 0; i < 3; ++i) {
        long long t1 = (long long)C * L;
        pad_convert_kernel<<<(unsigned)((t1 + 255) / 256), 256, 0, stream>>>(LAsrc[i], LAc[i], C, L, C, L);
        pad_convert_kernel<<<(unsigned)((t1 + 255) / 256), 256, 0, stream>>>(LBsrc[i], LBc[i], L, C, L, C);
        bias_fold_kernel<<<(L + 255) / 256, 256, 0, stream>>>(muL[i], LAsrc[i], nullptr, bL[i], C, L, L);
    }
    for (int i = 0; i < 4; ++i)
        bias_fold_kernel<<<(Ep + 255) / 256, 256, 0, stream>>>(muMLP[i], muMLP[i] ? W1f[i] : nullptr,
                                                               b1f[i], b1adj[i], C, E, Ep);
    fill_zero_kernel<<<(Ep + 255) / 256, 256, 0, stream>>>(zbias, Ep);

    // ---- r/k/v MLPs (WMMA)
    float* mlpOut[3] = { Rf, Kf, Vf };
    for (int i = 0; i < 3; ++i) {
        dim3 g1(Ep / BN, Mtok / BM);
        gemm_bf16_kernel<EPI_GELU_BF16><<<g1, 256, 0, stream>>>(Xbf, W1p[i], b1adj[i], H, Mtok, Ep, C);
        dim3 g2(C / BN, Mtok / BM);
        gemm_bf16_kernel<EPI_BIAS_F32><<<g2, 256, 0, stream>>>(H, W2p[i], b2f[i], mlpOut[i], Mtok, C, Ep);
    }

    // ---- LoRA chains: gate (sigmoid), iclr (sigmoid, +ab), decay (+db, exp(-c*sig(tanh)))
    {
        bf16_t* Hl = H;  // reuse as [Mtok][L]
        dim3 gs1(L / BN, Mtok / BM);
        dim3 gs2(C / BN, Mtok / BM);
        // gate
        gemm_bf16_kernel<EPI_ID_BF16><<<gs1, 256, 0, stream>>>(Xbf, LAc[0], bL[0], Hl, Mtok, L, C);
        gemm_bf16_kernel<EPI_SIG_F32><<<gs2, 256, 0, stream>>>(Hl, LBc[0], zbias, Gate, Mtok, C, L);
        // iclr
        gemm_bf16_kernel<EPI_ID_BF16><<<gs1, 256, 0, stream>>>(Xbf, LAc[1], bL[1], Hl, Mtok, L, C);
        gemm_bf16_kernel<EPI_SIG_F32><<<gs2, 256, 0, stream>>>(Hl, LBc[1], ab, Iclr, Mtok, C, L);
        // decay
        gemm_bf16_kernel<EPI_ID_BF16><<<gs1, 256, 0, stream>>>(Xbf, LAc[2], bL[2], Hl, Mtok, L, C);
        gemm_bf16_kernel<EPI_DECAY_F32><<<gs2, 256, 0, stream>>>(Hl, LBc[2], db, Decay, Mtok, C, L);
    }

    // ---- sequential state recurrence (includes kn normalize + bonus)
    recurrence_kernel<<<64, 64, 0, stream>>>(Rf, Kf, Vf, Iclr, Decay,
                                             rem_mult, iclr_mix, bonus_mult, RecO);

    // ---- LayerNorm * gate -> bf16 (into Xbf, which is free now)
    ln_gate_kernel<<<Mtok, 256, 0, stream>>>(RecO, Gate, ln_g, ln_b, Xbf);

    // ---- output MLP -> d_out (fp32)
    {
        dim3 g1(Ep / BN, Mtok / BM);
        gemm_bf16_kernel<EPI_GELU_BF16><<<g1, 256, 0, stream>>>(Xbf, W1p[3], b1adj[3], H, Mtok, Ep, C);
        dim3 g2(C / BN, Mtok / BM);
        gemm_bf16_kernel<EPI_BIAS_F32><<<g2, 256, 0, stream>>>(H, W2p[3], b2f[3], (float*)d_out, Mtok, C, Ep);
    }
}
